// WhisperEncoderAttention_7842610282964
// MI455X (gfx1250) — matrix operand order repaired and checked by reading, not compiled, not run on hardware
//
#include <hip/hip_runtime.h>
#include <hip/hip_bf16.h>

// ---------------------------------------------------------------------------
// WhisperEncoderAttention for MI455X (gfx1250, wave32, WMMA bf16->f32)
//   b=4, t=1500 (pad 1504), e=1280, h=20, d=64
//   LDS-cooperative tiling: GEMMs and attention share staged panels across
//   the 8 waves of each workgroup to cut L2 traffic ~8x vs per-wave tiling.
// ---------------------------------------------------------------------------

#define B_    4
#define T_    1500
#define TPAD  1504
#define E_    1280
#define H_    20
#define D_    64
#define NROW  (B_ * T_)          // 6000 rows of x
#define KSTEPS (E_ / 32)         // 40
#define QTILES (TPAD / 16)       // 94
#define KCHUNKS (TPAD / 32)      // 47
#define QKV_RB ((NROW + 127) / 128)   // 47 row blocks of 128
#define QKV_CB (E_ / 64)              // 20 col blocks of 64
#define OP_RB  QKV_RB                 // 47
#define OP_CB  (E_ / 128)             // 10 col blocks of 128
#define ATT_BLK ((QTILES + 7) / 8)    // 12 q-tile groups per (b,h)
#define SCALING 0.125f                // 64^-0.5

typedef __attribute__((ext_vector_type(16))) __bf16 v16bf;
typedef __attribute__((ext_vector_type(8)))  float  v8f;

union Frag {                      // 8 dwords == 16 bf16 == one WMMA A/B operand
    unsigned int u[8];
    v16bf        v;
};

__device__ __forceinline__ unsigned short f2bf(float f) {
    unsigned int u = __builtin_bit_cast(unsigned int, f);
    u = (u + 0x7FFFu + ((u >> 16) & 1u)) >> 16;   // round-to-nearest-even
    return (unsigned short)u;
}

__device__ __forceinline__ v8f wmma_bf16(const Frag& a, const Frag& b, v8f c) {
    return __builtin_amdgcn_wmma_f32_16x16x32_bf16(
        false, a.v, false, b.v, (short)0, c, false, false);
}

// A-fragment K offset for 16-bit A (16x32): VGPR i, half = lane/16
__device__ __forceinline__ int a_koff(int i, int half) {
    return (i < 4) ? (2 * i + 8 * half) : (16 + 2 * (i - 4) + 8 * half);
}
// B-fragment K offset for 16-bit B (32x16): VGPR j, half = lane/16
__device__ __forceinline__ int b_koff(int j, int half) {
    return (j < 4) ? (2 * j + 8 * half) : (16 + 2 * (j - 4) + 8 * half);
}

// ---------------------------------------------------------------------------
// 0) fp32 -> bf16 conversion
// ---------------------------------------------------------------------------
__global__ void cvt_bf16_kernel(const float* __restrict__ src,
                                unsigned short* __restrict__ dst, int n) {
    int i = blockIdx.x * blockDim.x + threadIdx.x;
    if (i < n) dst[i] = f2bf(src[i]);
}

// ---------------------------------------------------------------------------
// 0b) zero the T padding rows of q/k and pad cols of vT
// ---------------------------------------------------------------------------
__global__ void pad_zero_kernel(unsigned short* __restrict__ qb,
                                unsigned short* __restrict__ kb,
                                unsigned short* __restrict__ vtb) {
    int i = blockIdx.x * blockDim.x + threadIdx.x;   // B*H * 4 * 64 = 20480
    if (i >= B_ * H_ * 4 * D_) return;
    int d  = i & 63;
    int tp = (i >> 6) & 3;
    int bh = i >> 8;
    size_t idx = (((size_t)bh * TPAD) + (T_ + tp)) * D_ + d;
    qb[idx] = 0;
    kb[idx] = 0;
    size_t vidx = ((size_t)bh * D_ + d) * TPAD + (T_ + tp);
    vtb[vidx] = 0;
}

// ---------------------------------------------------------------------------
// 1) fused QKV projection.  Workgroup block tile: 128 rows x 64 cols (x3).
//    Per K-step(32): stage A 128x32 + three B 64x32 panels in LDS; each of
//    the 8 waves computes a 32x32 tile per matrix (12 WMMAs / step).
//    q -> [B,H,TPAD,D] bf16 (bias + *SCALING folded), k -> [B,H,TPAD,D],
//    v -> [B,H,D,TPAD] bf16 transposed (bias folded).
// ---------------------------------------------------------------------------
__global__ __launch_bounds__(256) void qkv_kernel(
    const unsigned short* __restrict__ xb,
    const unsigned short* __restrict__ wq,
    const unsigned short* __restrict__ wk,
    const unsigned short* __restrict__ wv,
    const float* __restrict__ bq,
    const float* __restrict__ bv,
    unsigned short* __restrict__ qb,
    unsigned short* __restrict__ kb,
    unsigned short* __restrict__ vtb) {

    __shared__ __align__(16) unsigned short Ash[128 * 32];      // 8 KB
    __shared__ __align__(16) unsigned short Bsh[3][64 * 32];    // 12 KB

    int tid  = threadIdx.x;
    int lw   = tid >> 5;
    int lane = tid & 31;
    int rb = blockIdx.x / QKV_CB;
    int cb = blockIdx.x % QKV_CB;
    int n = lane & 15, half = lane >> 4;
    int wy = lw >> 1, wx = lw & 1;         // 32-row group (0..3), 32-col group (0..1)
    const int rowbase = rb * 128;

    v8f acc[3][2][2] = {};

    for (int ks = 0; ks < KSTEPS; ++ks) {
        int k0 = ks * 32;
        // ---- stage A panel (128 x 32) : 2 x b128 per thread ----
#pragma unroll
        for (int s = 0; s < 2; ++s) {
            int o = tid * 16 + s * 8;
            int row = o >> 5, col = o & 31;
            int grow = rowbase + row;
            if (grow >= NROW) grow = NROW - 1;          // clamp (dup row, unstored)
            const uint4* src = (const uint4*)(xb + (size_t)grow * E_ + k0 + col);
            *(uint4*)(Ash + o) = *src;
            if (ks + 1 < KSTEPS)
                __builtin_prefetch((const void*)(xb + (size_t)grow * E_ + k0 + 32 + col), 0, 1);
        }
        // ---- stage B panels (3 x 64 x 32) : 1 x b128 per thread each ----
        {
            int o = tid * 8;
            int f = o >> 5, col = o & 31;
            size_t goff = (size_t)(cb * 64 + f) * E_ + k0 + col;
            *(uint4*)(Bsh[0] + o) = *(const uint4*)(wq + goff);
            *(uint4*)(Bsh[1] + o) = *(const uint4*)(wk + goff);
            *(uint4*)(Bsh[2] + o) = *(const uint4*)(wv + goff);
            if (ks + 1 < KSTEPS) {
                __builtin_prefetch((const void*)(wq + goff + 32), 0, 1);
                __builtin_prefetch((const void*)(wk + goff + 32), 0, 1);
                __builtin_prefetch((const void*)(wv + goff + 32), 0, 1);
            }
        }
        __syncthreads();

        // ---- per-wave fragments + 12 WMMAs ----
        Frag a[2];
#pragma unroll
        for (int mt = 0; mt < 2; ++mt) {
            const unsigned int* ap =
                (const unsigned int*)(Ash + (wy * 32 + mt * 16 + n) * 32);
#pragma unroll
            for (int i = 0; i < 8; ++i) a[mt].u[i] = ap[a_koff(i, half) >> 1];
        }
#pragma unroll
        for (int m = 0; m < 3; ++m) {
#pragma unroll
            for (int nt = 0; nt < 2; ++nt) {
                const unsigned int* bp =
                    (const unsigned int*)(Bsh[m] + (wx * 32 + nt * 16 + n) * 32);
                Frag bf;
#pragma unroll
                for (int j = 0; j < 8; ++j) bf.u[j] = bp[b_koff(j, half) >> 1];
                acc[m][0][nt] = wmma_bf16(a[0], bf, acc[m][0][nt]);
                acc[m][1][nt] = wmma_bf16(a[1], bf, acc[m][1][nt]);
            }
        }
        __syncthreads();
    }

    // ---- epilogue: bias, scale, scatter to q/k/vT ----
#pragma unroll
    for (int mt = 0; mt < 2; ++mt) {
#pragma unroll
        for (int nt = 0; nt < 2; ++nt) {
            int col = cb * 64 + wx * 32 + nt * 16 + n;
            int h = col >> 6, d = col & 63;
            float biasq = bq[col];
            float biasv = bv[col];
#pragma unroll
            for (int r = 0; r < 8; ++r) {
                int row = rowbase + wy * 32 + mt * 16 + r + 8 * half;
                if (row >= NROW) continue;
                int b = row / T_;
                int t = row - b * T_;
                size_t bh = (size_t)b * H_ + h;
                size_t qk = (bh * TPAD + t) * D_ + d;
                qb[qk] = f2bf((acc[0][mt][nt][r] + biasq) * SCALING);
                kb[qk] = f2bf(acc[1][mt][nt][r]);
                vtb[(bh * D_ + d) * TPAD + t] = f2bf(acc[2][mt][nt][r] + biasv);
            }
        }
    }
}

// ---------------------------------------------------------------------------
// 2) flash attention: one workgroup = 8 q-tiles of one (b,h).
//    Per 32-key chunk: stage K (32x64) and V^T (64x32) once in LDS, shared
//    by all 8 waves; P re-layout via per-wave LDS tile; online softmax.
// ---------------------------------------------------------------------------
__global__ __launch_bounds__(256) void attn_kernel(
    const unsigned short* __restrict__ qb,
    const unsigned short* __restrict__ kb,
    const unsigned short* __restrict__ vtb,
    unsigned short* __restrict__ attnb) {

    __shared__ __align__(16) unsigned short Ksh[32 * 64];       // 4 KB
    __shared__ __align__(16) unsigned short Vsh[64 * 32];       // 4 KB
    __shared__ __align__(16) unsigned short ptile[8][16 * 32];  // 8 KB

    int tid  = threadIdx.x;
    int lw   = tid >> 5;
    int lane = tid & 31;
    int bh  = blockIdx.x / ATT_BLK;
    int blk = blockIdx.x % ATT_BLK;
    int qt  = blk * 8 + lw;
    if (qt > QTILES - 1) qt = QTILES - 1;        // tail waves duplicate tile 93

    int n = lane & 15, half = lane >> 4;

    const unsigned short* qh = qb  + (size_t)bh * TPAD * D_;
    const unsigned short* kh = kb  + (size_t)bh * TPAD * D_;
    const unsigned short* vh = vtb + (size_t)bh * D_ * TPAD;

    // preload Q A-fragments (16x64 -> two 16x32 frags)
    Frag qa0, qa1;
    {
        const unsigned int* qr =
            (const unsigned int*)(qh + (size_t)(qt * 16 + n) * D_);
#pragma unroll
        for (int i = 0; i < 8; ++i) {
            int kd = a_koff(i, half);
            qa0.u[i] = qr[kd >> 1];
            qa1.u[i] = qr[(kd + 32) >> 1];
        }
    }

    float mrow[8], lrow[8];
    v8f o0 = {}, o1 = {}, o2 = {}, o3 = {};
#pragma unroll
    for (int r = 0; r < 8; ++r) { mrow[r] = -1e30f; lrow[r] = 0.f; }

    for (int kc = 0; kc < KCHUNKS; ++kc) {
        int kbase = kc * 32;

        // ---- stage K chunk (32 keys x 64 dims) and V^T chunk (64 x 32) ----
        {
            int o = tid * 8;
            int krow = o >> 6, kcol = o & 63;                 // 32 x 64
            *(uint4*)(Ksh + o) =
                *(const uint4*)(kh + (size_t)(kbase + krow) * D_ + kcol);
            int dcol = o >> 5, kk = o & 31;                   // 64 x 32
            *(uint4*)(Vsh + o) =
                *(const uint4*)(vh + (size_t)dcol * TPAD + kbase + kk);
            if (kc + 1 < KCHUNKS) {
                __builtin_prefetch(
                    (const void*)(kh + (size_t)(kbase + 32 + krow) * D_ + kcol), 0, 1);
                __builtin_prefetch(
                    (const void*)(vh + (size_t)dcol * TPAD + kbase + 32 + kk), 0, 1);
            }
        }
        __syncthreads();

        // ---- S = Q @ K^T for 32 keys (two 16x16 C tiles) ----
        v8f s[2];
#pragma unroll
        for (int nt = 0; nt < 2; ++nt) {
            const unsigned int* kp =
                (const unsigned int*)(Ksh + (nt * 16 + n) * 64);
            Frag b0, b1;
#pragma unroll
            for (int j = 0; j < 8; ++j) {
                int kd = b_koff(j, half);
                b0.u[j] = kp[kd >> 1];
                b1.u[j] = kp[(kd + 32) >> 1];
            }
            v8f c = {};
            c = wmma_bf16(qa0, b0, c);
            c = wmma_bf16(qa1, b1, c);
            if (kbase + nt * 16 + n >= T_) {                 // mask padded keys
#pragma unroll
                for (int r = 0; r < 8; ++r) c[r] = -1e30f;
            }
            s[nt] = c;
        }

        // ---- online softmax (reductions across the 16-lane half) ----
#pragma unroll
        for (int r = 0; r < 8; ++r) {
            float v = fmaxf(s[0][r], s[1][r]);
            v = fmaxf(v, __shfl_xor(v, 1));
            v = fmaxf(v, __shfl_xor(v, 2));
            v = fmaxf(v, __shfl_xor(v, 4));
            v = fmaxf(v, __shfl_xor(v, 8));
            float newm = fmaxf(mrow[r], v);
            float p0 = __expf(s[0][r] - newm);
            float p1 = __expf(s[1][r] - newm);
            s[0][r] = p0;
            s[1][r] = p1;
            float ps = p0 + p1;
            ps += __shfl_xor(ps, 1);
            ps += __shfl_xor(ps, 2);
            ps += __shfl_xor(ps, 4);
            ps += __shfl_xor(ps, 8);
            float alpha = __expf(mrow[r] - newm);
            lrow[r] = lrow[r] * alpha + ps;
            mrow[r] = newm;
            o0[r] *= alpha; o1[r] *= alpha; o2[r] *= alpha; o3[r] *= alpha;
        }

        // ---- C-layout P -> per-wave LDS tile (16x32 bf16) -> A-fragment ----
        unsigned short* pl = ptile[lw];
#pragma unroll
        for (int r = 0; r < 8; ++r) {
            int row = r + 8 * half;
            pl[row * 32 + n]      = f2bf(s[0][r]);
            pl[row * 32 + 16 + n] = f2bf(s[1][r]);
        }
        Frag pa;
        {
            const unsigned int* pr = (const unsigned int*)(pl + n * 32);
#pragma unroll
            for (int i = 0; i < 8; ++i) pa.u[i] = pr[a_koff(i, half) >> 1];
        }

        // ---- O += P @ V (four 16x16 d-tiles, V from LDS) ----
#pragma unroll
        for (int nt = 0; nt < 4; ++nt) {
            const unsigned int* vp =
                (const unsigned int*)(Vsh + (nt * 16 + n) * 32);
            Frag fv;
#pragma unroll
            for (int j = 0; j < 8; ++j) fv.u[j] = vp[b_koff(j, half) >> 1];
            if (nt == 0)      o0 = wmma_bf16(pa, fv, o0);
            else if (nt == 1) o1 = wmma_bf16(pa, fv, o1);
            else if (nt == 2) o2 = wmma_bf16(pa, fv, o2);
            else              o3 = wmma_bf16(pa, fv, o3);
        }
        __syncthreads();                  // protect Ksh/Vsh before next stage
    }

    // ---- epilogue: normalize, store attn output [B*T, E] bf16 ----
    int b = bh / H_, h = bh % H_;
#pragma unroll
    for (int r = 0; r < 8; ++r) {
        int t = qt * 16 + r + 8 * half;
        if (t >= T_) continue;
        float inv = 1.0f / lrow[r];
        size_t base = ((size_t)b * T_ + t) * E_ + h * D_;
        attnb[base + 0  + n] = f2bf(o0[r] * inv);
        attnb[base + 16 + n] = f2bf(o1[r] * inv);
        attnb[base + 32 + n] = f2bf(o2[r] * inv);
        attnb[base + 48 + n] = f2bf(o3[r] * inv);
    }
}

// ---------------------------------------------------------------------------
// 3) output projection: out = attn @ Wo^T + bo (fp32 store).
//    Block tile 128x128, per wave 32x64 (8 WMMAs / K-step).
// ---------------------------------------------------------------------------
__global__ __launch_bounds__(256) void oproj_kernel(
    const unsigned short* __restrict__ attnb,
    const unsigned short* __restrict__ wo,
    const float* __restrict__ bo,
    float* __restrict__ out) {

    __shared__ __align__(16) unsigned short Ash[128 * 32];   // 8 KB
    __shared__ __align__(16) unsigned short Bsh[128 * 32];   // 8 KB

    int tid  = threadIdx.x;
    int lw   = tid >> 5;
    int lane = tid & 31;
    int rb = blockIdx.x / OP_CB;
    int cb = blockIdx.x % OP_CB;
    int n = lane & 15, half = lane >> 4;
    int wy = lw >> 1, wx = lw & 1;        // 32-row group (0..3), 64-col group (0..1)
    const int rowbase = rb * 128;

    v8f acc[2][4] = {};

    for (int ks = 0; ks < KSTEPS; ++ks) {
        int k0 = ks * 32;
#pragma unroll
        for (int s = 0; s < 2; ++s) {
            int o = tid * 16 + s * 8;
            int row = o >> 5, col = o & 31;
            int grow = rowbase + row;
            if (grow >= NROW) grow = NROW - 1;
            *(uint4*)(Ash + o) =
                *(const uint4*)(attnb + (size_t)grow * E_ + k0 + col);
            *(uint4*)(Bsh + o) =
                *(const uint4*)(wo + (size_t)(cb * 128 + row) * E_ + k0 + col);
            if (ks + 1 < KSTEPS) {
                __builtin_prefetch(
                    (const void*)(attnb + (size_t)grow * E_ + k0 + 32 + col), 0, 1);
                __builtin_prefetch(
                    (const void*)(wo + (size_t)(cb * 128 + row) * E_ + k0 + 32 + col), 0, 1);
            }
        }
        __syncthreads();

        Frag a[2];
#pragma unroll
        for (int mt = 0; mt < 2; ++mt) {
            const unsigned int* ap =
                (const unsigned int*)(Ash + (wy * 32 + mt * 16 + n) * 32);
#pragma unroll
            for (int i = 0; i < 8; ++i) a[mt].u[i] = ap[a_koff(i, half) >> 1];
        }
#pragma unroll
        for (int nt = 0; nt < 4; ++nt) {
            const unsigned int* bp =
                (const unsigned int*)(Bsh + (wx * 64 + nt * 16 + n) * 32);
            Frag bf;
#pragma unroll
            for (int j = 0; j < 8; ++j) bf.u[j] = bp[b_koff(j, half) >> 1];
            acc[0][nt] = wmma_bf16(a[0], bf, acc[0][nt]);
            acc[1][nt] = wmma_bf16(a[1], bf, acc[1][nt]);
        }
        __syncthreads();
    }

#pragma unroll
    for (int mt = 0; mt < 2; ++mt) {
#pragma unroll
        for (int nt = 0; nt < 4; ++nt) {
            int col = cb * 128 + wx * 64 + nt * 16 + n;
            float bias = bo[col];
#pragma unroll
            for (int r = 0; r < 8; ++r) {
                int row = rowbase + wy * 32 + mt * 16 + r + 8 * half;
                if (row >= NROW) continue;
                out[(size_t)row * E_ + col] = acc[mt][nt][r] + bias;
            }
        }
    }
}

// ---------------------------------------------------------------------------
// launch
// ---------------------------------------------------------------------------
extern "C" void kernel_launch(void* const* d_in, const int* in_sizes, int n_in,
                              void* d_out, int out_size, void* d_ws, size_t ws_size,
                              hipStream_t stream) {
    const float* x  = (const float*)d_in[0];
    const float* Wq = (const float*)d_in[1];
    const float* bq = (const float*)d_in[2];
    const float* Wk = (const float*)d_in[3];
    const float* Wv = (const float*)d_in[4];
    const float* bv = (const float*)d_in[5];
    const float* Wo = (const float*)d_in[6];
    const float* bo = (const float*)d_in[7];
    float* out = (float*)d_out;

    // workspace layout (bytes, all 256-aligned)
    char* ws = (char*)d_ws;
    const size_t XB   = (size_t)NROW * E_ * 2;           // 15,360,000
    const size_t WB   = (size_t)E_ * E_ * 2;             //  3,276,800
    const size_t QKVB = (size_t)B_ * H_ * TPAD * D_ * 2; // 15,400,960
    size_t off = 0;
    unsigned short* xb    = (unsigned short*)(ws + off); off += XB;
    unsigned short* wqb   = (unsigned short*)(ws + off); off += WB;
    unsigned short* wkb   = (unsigned short*)(ws + off); off += WB;
    unsigned short* wvb   = (unsigned short*)(ws + off); off += WB;
    unsigned short* wob   = (unsigned short*)(ws + off); off += WB;
    unsigned short* qbuf  = (unsigned short*)(ws + off); off += QKVB;
    unsigned short* kbuf  = (unsigned short*)(ws + off); off += QKVB;
    unsigned short* vtbuf = (unsigned short*)(ws + off); off += QKVB;
    unsigned short* attnb = (unsigned short*)(ws + off); off += XB;
    (void)ws_size; (void)in_sizes; (void)n_in; (void)out_size;

    const int XN = NROW * E_;      // 7,680,000
    const int WN = E_ * E_;        // 1,638,400
    cvt_bf16_kernel<<<(XN + 255) / 256, 256, 0, stream>>>(x,  xb,  XN);
    cvt_bf16_kernel<<<(WN + 255) / 256, 256, 0, stream>>>(Wq, wqb, WN);
    cvt_bf16_kernel<<<(WN + 255) / 256, 256, 0, stream>>>(Wk, wkb, WN);
    cvt_bf16_kernel<<<(WN + 255) / 256, 256, 0, stream>>>(Wv, wvb, WN);
    cvt_bf16_kernel<<<(WN + 255) / 256, 256, 0, stream>>>(Wo, wob, WN);

    pad_zero_kernel<<<(B_ * H_ * 4 * D_ + 255) / 256, 256, 0, stream>>>(
        qbuf, kbuf, vtbuf);

    qkv_kernel<<<QKV_RB * QKV_CB, 256, 0, stream>>>(
        xb, wqb, wkb, wvb, bq, bv, qbuf, kbuf, vtbuf);

    attn_kernel<<<B_ * H_ * ATT_BLK, 256, 0, stream>>>(
        qbuf, kbuf, vtbuf, attnb);

    oproj_kernel<<<OP_RB * OP_CB, 256, 0, stream>>>(
        attnb, wob, bo, out);
}
